// MambaRecurrentFusion_14912126452379
// MI455X (gfx1250) — matrix operand order repaired and checked by reading, not compiled, not run on hardware
//
#include <hip/hip_runtime.h>

#define B_N 4096
#define E_N 1536
#define D_N 3072
#define NS  5
#define CAP 4736      // 4096 + 5*128 padding capacity, multiple of 128
#define MTILES 37     // CAP / 128
#define PITCH 48      // LDS row pitch in bf16 elems (96B: 16B-aligned, conflict-light)

typedef __attribute__((ext_vector_type(16))) __bf16 v16bf;
typedef __attribute__((ext_vector_type(8)))  __bf16 v8bf;
typedef __attribute__((ext_vector_type(8)))  float  v8f;

// native converts -> v_cvt_(pk_)bf16_f32 instead of manual RNE bit-twiddling
__device__ __forceinline__ unsigned short f2bf(float f) {
  __bf16 h = (__bf16)f;
  unsigned short u;
  __builtin_memcpy(&u, &h, 2);
  return u;
}
__device__ __forceinline__ unsigned pk2bf(float a, float b) {
  return (unsigned)f2bf(a) | ((unsigned)f2bf(b) << 16);
}

__device__ __forceinline__ v16bf load_frag(const unsigned short* p0,
                                           const unsigned short* p1) {
  v16bf r;
  ((v8bf*)&r)[0] = *(const v8bf*)p0;   // ds_load_b128
  ((v8bf*)&r)[1] = *(const v8bf*)p1;   // ds_load_b128
  return r;
}

__device__ __forceinline__ v8f wmma_bf16(v16bf a, v16bf b, v8f c) {
  return __builtin_amdgcn_wmma_f32_16x16x32_bf16(
      false, a, false, b, (short)0, c, false, false);
}

// ---------------------------------------------------------------------------
// K1: selector logits + argmax + fp32->bf16 conversion of x = [ch|pe]
// ---------------------------------------------------------------------------
__global__ __launch_bounds__(256)
void k_sel(const float* __restrict__ ch, const float* __restrict__ pe,
           const float* __restrict__ Wsel, const float* __restrict__ bsel,
           unsigned short* __restrict__ xbf, int* __restrict__ idx) {
  __shared__ float red[NS][256];
  const int b = blockIdx.x;
  const int tid = threadIdx.x;
  float acc[NS] = {0.f, 0.f, 0.f, 0.f, 0.f};
  for (int k = tid; k < D_N; k += 256) {
    float xv = (k < E_N) ? ch[(size_t)b * E_N + k]
                         : pe[(size_t)b * E_N + (k - E_N)];
    xbf[(size_t)b * D_N + k] = f2bf(xv);
#pragma unroll
    for (int e = 0; e < NS; ++e) acc[e] += xv * Wsel[(size_t)e * D_N + k];
  }
#pragma unroll
  for (int e = 0; e < NS; ++e) red[e][tid] = acc[e];
  __syncthreads();
  for (int s = 128; s > 0; s >>= 1) {
    if (tid < s) {
#pragma unroll
      for (int e = 0; e < NS; ++e) red[e][tid] += red[e][tid + s];
    }
    __syncthreads();
  }
  if (tid == 0) {
    float best = red[0][0] + bsel[0];
    int bi = 0;
    for (int e = 1; e < NS; ++e) {
      float v = red[e][0] + bsel[e];
      if (v > best) { best = v; bi = e; }   // strict > == jnp.argmax tie rule
    }
    idx[b] = bi;
  }
}

// ---------------------------------------------------------------------------
// K2: padded per-expert permutation (mult of 128) + 128-row tile->expert map
// ---------------------------------------------------------------------------
__global__ __launch_bounds__(1024)
void k_group(const int* __restrict__ idx, int* __restrict__ perm,
             int* __restrict__ tile_e, int* __restrict__ hdr) {
  __shared__ int cnt[NS], cur[NS], soff[NS + 1];
  const int tid = threadIdx.x;
  if (tid < NS) { cnt[tid] = 0; cur[tid] = 0; }
  __syncthreads();
  for (int b = tid; b < B_N; b += 1024) atomicAdd(&cnt[idx[b]], 1);
  __syncthreads();
  if (tid == 0) {
    int run = 0;
    for (int e = 0; e < NS; ++e) {
      soff[e] = run;
      run += ((cnt[e] + 127) >> 7) << 7;   // pad each group to 128 rows
    }
    soff[NS] = run;
    hdr[0] = run;
    for (int e = 0; e <= NS; ++e) hdr[1 + e] = soff[e];
  }
  __syncthreads();
  for (int p = tid; p < CAP; p += 1024) perm[p] = -1;
  __syncthreads();
  for (int b = tid; b < B_N; b += 1024) {
    int e = idx[b];
    int pos = soff[e] + atomicAdd(&cur[e], 1);
    perm[pos] = b;
  }
  __syncthreads();
  for (int t = tid; t < MTILES; t += 1024) {
    int base = t * 128, te = -1;
#pragma unroll
    for (int e = 0; e < NS; ++e)
      if (base >= soff[e] && base < soff[e + 1]) te = e;
    tile_e[t] = te;
  }
}

// ---------------------------------------------------------------------------
// K3/K4: grouped expert GEMM, block tile 128x128, K step 32.
// 8 waves (4 Mx2 N): each wave 32x64 via 8 WMMAs / K step.
// MODE==1: h = relu(gather(x) @ A[e]);  MODE==2: y = h @ C[e] scattered back.
// ---------------------------------------------------------------------------
template <int MODE>
__global__ __launch_bounds__(256)
void k_expert_gemm(const unsigned short* __restrict__ in,
                   const float* __restrict__ W,
                   unsigned short* __restrict__ outp,
                   const int* __restrict__ perm,
                   const int* __restrict__ tile_e) {
  const int mt = blockIdx.x;
  const int e = tile_e[mt];
  if (e < 0) return;
  const float* We = W + (size_t)e * D_N * D_N;
  const int m0 = mt * 128;
  const int n0 = blockIdx.y * 128;

  __shared__ unsigned short sA[128][PITCH];  // rows x K
  __shared__ unsigned short sB[128][PITCH];  // N-major: sB[n][k]
  __shared__ int srow[128];                  // input-row map
  __shared__ int smap[128];                  // output-row map

  const int tid = threadIdx.x;
  if (tid < 128) {
    int p = m0 + tid;
    int pv = perm[p];
    srow[tid] = (MODE == 1) ? pv : p;
    smap[tid] = (MODE == 1) ? p : pv;
  }

  const int lane = tid & 31, wid = tid >> 5;
  const int wm = wid >> 1, wn = wid & 1;     // 4 M-waves x 2 N-waves
  const int half = lane >> 4, mr = lane & 15;

  v8f acc[2][4];
#pragma unroll
  for (int m = 0; m < 2; ++m)
#pragma unroll
    for (int s = 0; s < 4; ++s)
#pragma unroll
      for (int j = 0; j < 8; ++j) acc[m][s][j] = 0.f;

  for (int k0 = 0; k0 < D_N; k0 += 32) {
    __syncthreads();
    // stage A: 128 rows x 32 bf16 (dword copies, zeros for pad rows)
#pragma unroll
    for (int i = 0; i < 8; ++i) {
      int id = tid + i * 256;
      int r = id >> 4, c2 = id & 15;
      int gr = srow[r];
      unsigned w = 0u;
      if (gr >= 0)
        w = *(const unsigned*)(in + (size_t)gr * D_N + (k0 + c2 * 2));
      *(unsigned*)&sA[r][c2 * 2] = w;
    }
    // stage B transposed: sB[n][k] = bf16(We[k0+k][n0+n]); float4 reads +
    // prefetch of next K-slab (global_prefetch_b8) to hide fp32 fetch latency
    {
      int c = (tid & 31) << 2;        // 0..124
      int kb = tid >> 5;              // 0..7
#pragma unroll
      for (int p = 0; p < 4; ++p) {
        int k = kb + (p << 3);        // 0..31
        const float* src = We + (size_t)(k0 + k) * D_N + (n0 + c);
        const float4 f = *(const float4*)src;
        if (k0 + 32 < D_N)
          __builtin_prefetch(src + (size_t)32 * D_N, 0, 3);
        sB[c + 0][k] = f2bf(f.x);
        sB[c + 1][k] = f2bf(f.y);
        sB[c + 2][k] = f2bf(f.z);
        sB[c + 3][k] = f2bf(f.w);
      }
    }
    __syncthreads();

    v16bf a0 = load_frag(&sA[wm * 32 + mr][half * 8],
                         &sA[wm * 32 + mr][16 + half * 8]);
    v16bf a1 = load_frag(&sA[wm * 32 + 16 + mr][half * 8],
                         &sA[wm * 32 + 16 + mr][16 + half * 8]);
#pragma unroll
    for (int s = 0; s < 4; ++s) {
      const unsigned short* bp = &sB[wn * 64 + s * 16 + mr][half * 8];
      v16bf bb = load_frag(bp, bp + 16);
      acc[0][s] = wmma_bf16(a0, bb, acc[0][s]);
      acc[1][s] = wmma_bf16(a1, bb, acc[1][s]);
    }
  }

  // epilogue: (optional relu) -> bf16 store; row j of VGPR = j + 8*(lane>=16)
#pragma unroll
  for (int m = 0; m < 2; ++m) {
#pragma unroll
    for (int j = 0; j < 8; ++j) {
      int rloc = wm * 32 + m * 16 + j + half * 8;
      int orow = smap[rloc];
      if (orow < 0) continue;
#pragma unroll
      for (int s = 0; s < 4; ++s) {
        float v = acc[m][s][j];
        if (MODE == 1) v = v > 0.f ? v : 0.f;
        outp[(size_t)orow * D_N + (n0 + wn * 64 + s * 16 + mr)] = f2bf(v);
      }
    }
  }
}

// ---------------------------------------------------------------------------
// K5: fused gi = y @ w_ih^T (+biases) + GRU gates + residual -> d_out [B,E]
// Block: 64 rows x same 64 cols of all 3 gates. 8 waves (2 M x 4 N):
// wave = 32 rows x 16 cols x 3 gates = 6 WMMAs / K step, epilogue in-register.
// ---------------------------------------------------------------------------
__global__ __launch_bounds__(256)
void k_gru(const unsigned short* __restrict__ y, const float* __restrict__ wih,
           const float* __restrict__ bih, const float* __restrict__ bhh,
           const float* __restrict__ ch, const float* __restrict__ pe,
           float* __restrict__ outp) {
  __shared__ unsigned short sA[64][PITCH];
  __shared__ unsigned short sB[192][PITCH];   // 3 gates x 64 cols, N-major
  const int m0 = blockIdx.x * 64;
  const int c0 = blockIdx.y * 64;
  const int tid = threadIdx.x;
  const int lane = tid & 31, wid = tid >> 5;
  const int wm = wid >> 2, wn = wid & 3;      // 2 M-waves x 4 N-waves
  const int half = lane >> 4, mr = lane & 15;

  v8f acc[3][2];
#pragma unroll
  for (int g = 0; g < 3; ++g)
#pragma unroll
    for (int m = 0; m < 2; ++m)
#pragma unroll
      for (int j = 0; j < 8; ++j) acc[g][m][j] = 0.f;

  for (int k0 = 0; k0 < D_N; k0 += 32) {
    __syncthreads();
#pragma unroll
    for (int i = 0; i < 4; ++i) {
      int id = tid + i * 256;
      int r = id >> 4, c2 = id & 15;
      *(unsigned*)&sA[r][c2 * 2] =
          *(const unsigned*)(y + (size_t)(m0 + r) * D_N + (k0 + c2 * 2));
    }
    {
      int r8 = tid >> 3;            // 0..31
      int c = (tid & 7) << 2;       // 0..28
#pragma unroll
      for (int p = 0; p < 6; ++p) {
        int ri = r8 + p * 32;       // 0..191  (gate g = ri/64, col n = ri%64)
        int g = ri >> 6, n = ri & 63;
        const float* src = wih + (size_t)(g * E_N + c0 + n) * D_N + (k0 + c);
        const float4 f = *(const float4*)src;
        if (k0 + 32 < D_N) __builtin_prefetch(src + 32, 0, 3);
        *(unsigned*)&sB[ri][c]     = pk2bf(f.x, f.y);
        *(unsigned*)&sB[ri][c + 2] = pk2bf(f.z, f.w);
      }
    }
    __syncthreads();

    v16bf a0 = load_frag(&sA[wm * 32 + mr][half * 8],
                         &sA[wm * 32 + mr][16 + half * 8]);
    v16bf a1 = load_frag(&sA[wm * 32 + 16 + mr][half * 8],
                         &sA[wm * 32 + 16 + mr][16 + half * 8]);
#pragma unroll
    for (int g = 0; g < 3; ++g) {
      const unsigned short* bp = &sB[g * 64 + wn * 16 + mr][half * 8];
      v16bf bb = load_frag(bp, bp + 16);
      acc[g][0] = wmma_bf16(a0, bb, acc[g][0]);
      acc[g][1] = wmma_bf16(a1, bb, acc[g][1]);
    }
  }

  const int col = c0 + wn * 16 + mr;          // fixed per lane
  const float bir = bih[col] + bhh[col];
  const float biz = bih[E_N + col] + bhh[E_N + col];
  const float bin = bih[2 * E_N + col];
  const float bhn = bhh[2 * E_N + col];
#pragma unroll
  for (int m = 0; m < 2; ++m) {
#pragma unroll
    for (int j = 0; j < 8; ++j) {
      int row = m0 + wm * 32 + m * 16 + j + half * 8;
      float r = 1.f / (1.f + expf(-(acc[0][m][j] + bir)));
      float z = 1.f / (1.f + expf(-(acc[1][m][j] + biz)));
      float n = tanhf(acc[2][m][j] + bin + r * bhn);
      float hn = (1.f - z) * n;
      size_t o = (size_t)row * E_N + col;
      outp[o] = hn + ch[o] + pe[o];
    }
  }
}

// ---------------------------------------------------------------------------
extern "C" void kernel_launch(void* const* d_in, const int* in_sizes, int n_in,
                              void* d_out, int out_size, void* d_ws,
                              size_t ws_size, hipStream_t stream) {
  const float* ch   = (const float*)d_in[0];
  const float* pe   = (const float*)d_in[1];
  const float* A    = (const float*)d_in[2];
  const float* C    = (const float*)d_in[3];
  const float* Wsel = (const float*)d_in[4];
  const float* bsel = (const float*)d_in[5];
  const float* wih  = (const float*)d_in[6];
  // d_in[7] = w_hh (unused: h0 == 0)
  const float* bih  = (const float*)d_in[8];
  const float* bhh  = (const float*)d_in[9];
  float* out = (float*)d_out;

  char* ws = (char*)d_ws;
  constexpr size_t OFF_HDR  = 0;
  constexpr size_t OFF_TE   = 128;                                 // 37 ints
  constexpr size_t OFF_IDX  = 768;
  constexpr size_t OFF_PERM = 17152;                               // 4736 ints
  constexpr size_t OFF_X    = 36096;                               // 256-aligned
  constexpr size_t OFF_H    = OFF_X + (size_t)B_N * D_N * 2;       // 25,201,920
  constexpr size_t OFF_Y    = OFF_H + (size_t)CAP * D_N * 2;       // 54,299,904

  int* hdr    = (int*)(ws + OFF_HDR);
  int* tile_e = (int*)(ws + OFF_TE);
  int* idx    = (int*)(ws + OFF_IDX);
  int* perm   = (int*)(ws + OFF_PERM);
  unsigned short* xbf = (unsigned short*)(ws + OFF_X);
  unsigned short* hbf = (unsigned short*)(ws + OFF_H);
  unsigned short* ybf = (unsigned short*)(ws + OFF_Y);

  k_sel<<<B_N, 256, 0, stream>>>(ch, pe, Wsel, bsel, xbf, idx);
  k_group<<<1, 1024, 0, stream>>>(idx, perm, tile_e, hdr);
  k_expert_gemm<1><<<dim3(MTILES, D_N / 128), 256, 0, stream>>>(
      xbf, A, hbf, perm, tile_e);
  k_expert_gemm<2><<<dim3(MTILES, D_N / 128), 256, 0, stream>>>(
      hbf, C, ybf, perm, tile_e);
  k_gru<<<dim3(B_N / 64, E_N / 64), 256, 0, stream>>>(
      ybf, wih, bih, bhh, ch, pe, out);
}
